// Assistance_69758858821971
// MI455X (gfx1250) — matrix operand order repaired and checked by reading, not compiled, not run on hardware
//
#include <hip/hip_runtime.h>
#include <hip/hip_bf16.h>
#include <math.h>

// CDNA5 wave32 WMMA types
typedef __attribute__((ext_vector_type(16))) _Float16 v16h;
typedef __attribute__((ext_vector_type(8)))  _Float16 v8h;
typedef __attribute__((ext_vector_type(8)))  float    v8f;

#define RR     21
#define NPTS   441            // 21*21
#define KPAD   448            // 14 chunks * 32
#define NCHUNK 14
#define HPDIM  64
#define HIMG   147
#define ETA_   0.01f
#define TAU_   0.1f
#define PIF    3.14159265358979323846f

__device__ __forceinline__ float mod2pi(float x) {
    float m = fmodf(x, 2.0f * PIF);
    return (m < 0.0f) ? m + 2.0f * PIF : m;
}

// g(dtheta) = (dtheta/pi - 1)^35 via square-and-multiply (u^32 * u^2 * u)
__device__ __forceinline__ float gpow35(float dtheta) {
    float u  = dtheta * (1.0f / PIF) - 1.0f;
    float u2 = u * u, u4 = u2 * u2, u8 = u4 * u4, u16 = u8 * u8, u32 = u16 * u16;
    return u32 * u2 * u;
}

// 64 threads = 2 waves; each wave owns 4 output pixels -> 8 px/block, 512 blocks.
__global__ __launch_bounds__(64) void wedge_wmma_kernel(
    const float* __restrict__ ests,     // (4096, 5)
    const float* __restrict__ noisy,    // (147, 147, 3) NHWC
    float* __restrict__ out)            // (3, 3, 64, 64)
{
    // Both staged in the WMMA K-padded layout: [q][row/k][t], t in [0,448)
    //   patch rows: 0..2 = channels (f16), 3 = ones (t<441) -> wsum row
    //   wedge rows: 0..2 = wedge indicators, 3 = zeros -> padding column
    __shared__ alignas(32) _Float16 patch_lds[2][4 * 4 * KPAD];   // 2*14336 B
    __shared__ alignas(32) _Float16 wedge_lds[2][4 * 4 * KPAD];   // 2*14336 B

    const int lane    = threadIdx.x & 31;
    const int wave    = threadIdx.x >> 5;
    const int waveg   = blockIdx.x * 2 + wave;   // 0..1023
    const int pixbase = waveg * 4;               // 4 pixels per wave

    // warm the patch footprint in WGP$ (4 pixels are w-adjacent -> same rows)
    {
        const int h0 = (pixbase >> 6) * 2;
        const int w0 = (pixbase & 63) * 2;
        for (int r = lane; r < RR; r += 32) {
            __builtin_prefetch(&noisy[((h0 + r) * HIMG + w0) * 3], 0, 3);
        }
    }

    // ---- Phase 1: wedge field (transcendental-bound) + f16 patch staging ----
    for (int q = 0; q < 4; ++q) {
        const int pix = pixbase + q;
        const int yy0 = (pix >> 6) * 2;
        const int xx0 = (pix & 63) * 2;
        const float* e = ests + pix * 5;
        float ang0 = mod2pi((e[0] + 1.0f) * PIF);
        float ang1 = mod2pi((e[1] + 1.0f) * PIF);
        float ang2 = mod2pi((e[2] + 1.0f) * PIF);
        float lo = fminf(ang0, ang1), hi = fmaxf(ang0, ang1);
        float a1 = fminf(lo, ang2);
        float a3 = fmaxf(hi, ang2);
        float a2 = ang0 + ang1 + ang2 - a1 - a3;
        const float x0 = e[3] * 3.0f, y0 = e[4] * 3.0f;
        const float a4 = 0.5f * (a1 + a3) + ((mod2pi(0.5f * (a1 - a3)) >= PIF) ? PIF : 0.0f);

        float s1, c1, s2, c2, s3, c3, s4, c4;
        sincosf(a1, &s1, &c1);
        sincosf(a2, &s2, &c2);
        sincosf(a3, &s3, &c3);
        sincosf(a4, &s4, &c4);

        const float m42   = mod2pi(a2 - a4);
        const float m13   = mod2pi(a3 - a1);
        const float sgn42 = (m42 < PIF) ? 1.0f : -1.0f;
        const float sgn13 = (m13 < PIF) ? 1.0f : -1.0f;
        const float g42   = gpow35(m42) * TAU_;
        const float g13   = gpow35(m13) * TAU_;

        _Float16* wq = &wedge_lds[wave][(q * 4) * KPAD];
        _Float16* pq = &patch_lds[wave][(q * 4) * KPAD];

        for (int t = lane; t < KPAD; t += 32) {   // 14 iters/lane
            float w0v = 0.0f, w1v = 0.0f, w2v = 0.0f;
            float p0 = 0.0f, p1 = 0.0f, p2 = 0.0f, ones = 0.0f;
            if (t < NPTS) {
                const int   r  = t / RR;
                const int   s  = t - r * RR;
                // image patch element (3 channels contiguous in NHWC)
                const float* px = &noisy[((yy0 + r) * HIMG + (xx0 + s)) * 3];
                p0 = px[0]; p1 = px[1]; p2 = px[2]; ones = 1.0f;
                // wedge indicators
                const float x  = -1.0f + 0.1f * (float)s;   // linspace(-1,1,21)
                const float y  = -1.0f + 0.1f * (float)r;
                const float dx = x - x0, dy = y - y0;
                const float l1 = -s1 * dx + c1 * dy;
                const float l2 = -s2 * dx + c2 * dy;
                const float l3 = -s3 * dx + c3 * dy;
                const float l4 = -s4 * dx + c4 * dy;
                const float d13 = sgn13 * fminf(sgn13 * l1, -sgn13 * l3) + g13;
                const float d42 = sgn42 * fminf(sgn42 * l4, -sgn42 * l2) + g42;
                const float h0v = 0.5f + (1.0f / PIF) * atanf(d13 * (1.0f / ETA_));
                const float h1v = 0.5f + (1.0f / PIF) * atanf(d42 * (1.0f / ETA_));
                w0v = 1.0f - h0v;
                w1v = h0v * (1.0f - h1v);
                w2v = h0v * h1v;
            }
            wq[0 * KPAD + t] = (_Float16)w0v;
            wq[1 * KPAD + t] = (_Float16)w1v;
            wq[2 * KPAD + t] = (_Float16)w2v;
            wq[3 * KPAD + t] = (_Float16)0.0f;     // padding column
            pq[0 * KPAD + t] = (_Float16)p0;
            pq[1 * KPAD + t] = (_Float16)p1;
            pq[2 * KPAD + t] = (_Float16)p2;
            pq[3 * KPAD + t] = (_Float16)ones;     // wsum row
        }
    }
    __syncthreads();

    // ---- Phase 2: branch-free LDS -> WMMA fragments ----
    // A (16x32 f16): lane L<16 -> M=L, K in {0..15}; L>=16 -> K in {16..31}
    // B (32x16 f16): lane L<16 -> N=L, K in {0..15}; L>=16 -> K in {16..31}
    const int mrow   = lane & 15;
    const int is_hi  = lane >> 4;
    const int pA     = mrow >> 2;          // pixel slot for A row
    const int rowA   = mrow & 3;           // 0..2 channel, 3 ones
    const int kbaseA = is_hi ? 16 : 0;
    const int qB     = (lane & 15) >> 2;   // pixel slot for B col
    const int kB     = lane & 3;           // wedge idx, 3 = zero pad
    const int kbaseB = is_hi ? 16 : 0;

    const _Float16* arow = &patch_lds[wave][(pA * 4 + rowA) * KPAD];
    const _Float16* brow = &wedge_lds[wave][(qB * 4 + kB) * KPAD];

    v8f acc = {};
#pragma unroll
    for (int ch = 0; ch < NCHUNK; ++ch) {
        const int tb = ch * 32;
        const v8h a_lo = *(const v8h*)&arow[tb + kbaseA];        // 16B aligned
        const v8h a_hi = *(const v8h*)&arow[tb + kbaseA + 8];    // 16B aligned
        const v16h a = __builtin_shufflevector(a_lo, a_hi,
            0, 1, 2, 3, 4, 5, 6, 7, 8, 9, 10, 11, 12, 13, 14, 15);
        const v16h b = *(const v16h*)&brow[tb + kbaseB];         // 32B aligned
        acc = __builtin_amdgcn_wmma_f32_16x16x32_f16(
            /*neg_a=*/false, a, /*neg_b=*/false, b,
            /*c_mod=*/(short)0, acc, /*reuse_a=*/false, /*reuse_b=*/false);
    }

    // ---- Extract block-diagonal 4x3 result blocks (p == q) ----
    // D layout: VGPR j, lanes 0..15 -> M=j, N=lane; lanes 16..31 -> M=8+j, N=lane-16
    if (kB < 3 && (qB >> 1) == is_hi) {
        float n0, n1, n2, ws;
        if (qB & 1) { n0 = acc[4]; n1 = acc[5]; n2 = acc[6]; ws = acc[7]; }
        else        { n0 = acc[0]; n1 = acc[1]; n2 = acc[2]; ws = acc[3]; }
        ws += 1e-10f;
        const float inv = 1.0f / ws;
        const int pix = pixbase + qB;
        const int h = pix >> 6, w = pix & 63;
        out[((0 * 3 + kB) * HPDIM + h) * HPDIM + w] = n0 * inv;
        out[((1 * 3 + kB) * HPDIM + h) * HPDIM + w] = n1 * inv;
        out[((2 * 3 + kB) * HPDIM + h) * HPDIM + w] = n2 * inv;
    }
}

extern "C" void kernel_launch(void* const* d_in, const int* in_sizes, int n_in,
                              void* d_out, int out_size, void* d_ws, size_t ws_size,
                              hipStream_t stream) {
    (void)in_sizes; (void)n_in; (void)out_size; (void)d_ws; (void)ws_size;
    const float* ests  = (const float*)d_in[0];   // (1, 4096, 5)
    const float* noisy = (const float*)d_in[1];   // (1, 147, 147, 3)
    // d_in[2] = gt_image, d_in[3] = alpha : unused by the reference output
    float* out = (float*)d_out;                   // (1, 3, 3, 64, 64)

    // 4096 pixels / (2 waves * 4 px per wave) = 512 blocks of 64 threads
    wedge_wmma_kernel<<<dim3(512), dim3(64), 0, stream>>>(ests, noisy, out);
}
